// Mamba2_44229573214748
// MI455X (gfx1250) — compile-verified, matrix operand order checked by reading, not run on hardware
//
#include <hip/hip_runtime.h>
#include <hip/hip_bf16.h>

// ---- problem constants ------------------------------------------------------
#define BD    2
#define LD    2048
#define DM    1024
#define DI    2048
#define DSZ   128
#define NH    32
#define HD    64
#define CK    64
#define NCH   (LD / CK)              // 32 chunks per batch row
#define BL    (BD * LD)              // 4096 token rows
#define DPROJ (2 * DI + 2 * DSZ + NH) // 4384
#define CONVD (DI + 2 * DSZ)         // 2304

typedef __attribute__((ext_vector_type(16))) __bf16 v16bf;
typedef __attribute__((ext_vector_type(8)))  float  v8f;

// ---- WMMA helpers (wave32, 16x16x32 bf16 -> f32) ---------------------------
__device__ __forceinline__ v8f wmma_bf16(v16bf a, v16bf b, v8f c) {
  return __builtin_amdgcn_wmma_f32_16x16x32_bf16(false, a, false, b, (short)0, c,
                                                 false, false);
}

// Fragment with 16 rows x 32 K: lane holds row = lane%16, K off = (lane/16)*16,
// 16 contiguous bf16. Same mapping used for A (MxK) and B-operand (NxK = B^T),
// so per-slot K pairing between A and B is consistent -> correct dot product.
__device__ __forceinline__ v16bf frag_ld(const __bf16* p, int ld, int lane) {
  const __bf16* q = p + (lane & 15) * ld + (lane >> 4) * 16;
  v16bf f;
#pragma unroll
  for (int i = 0; i < 16; ++i) f[i] = q[i];
  return f;
}

// Same fragment but from fp32 source (converted on the fly).
__device__ __forceinline__ v16bf frag_ld_f32(const float* p, int ld, int lane) {
  const float* q = p + (lane & 15) * ld + (lane >> 4) * 16;
  v16bf f;
#pragma unroll
  for (int i = 0; i < 16; ++i) f[i] = (__bf16)q[i];
  return f;
}

// B-operand fragment for a (K x N) matrix stored K-strided: element (k,n) at
// src[k*ldk + n]; each K element scaled by scale[k] (dt or dt*decay).
__device__ __forceinline__ v16bf frag_ld_xk(const float* src, int ldk,
                                            const float* scale, int lane) {
  const int n  = lane & 15;
  const int k0 = (lane >> 4) * 16;
  v16bf f;
#pragma unroll
  for (int i = 0; i < 16; ++i)
    f[i] = (__bf16)(src[(k0 + i) * ldk + n] * scale[k0 + i]);
  return f;
}

// ---- CDNA5 async global->LDS copy (ASYNCcnt path, bypasses VGPRs) ----------
__device__ __forceinline__ void async_ld_b128(unsigned lds_off,
                                              const void* gptr) {
  asm volatile("global_load_async_to_lds_b128 %0, %1, off"
               :: "v"(lds_off), "v"(gptr)
               : "memory");
}
__device__ __forceinline__ void wait_asynccnt0() {
  asm volatile("s_wait_asynccnt 0" ::: "memory");
}
// LDS byte offset of a shared-memory pointer: generic LDS addresses carry the
// wave-relative LDS offset in addr[31:0] (ISA aperture rule).
__device__ __forceinline__ unsigned lds_off_of(const void* p) {
  return (unsigned)(size_t)p;
}

// ---- fp32 -> bf16 convert ---------------------------------------------------
__global__ void __launch_bounds__(256)
k_f2bf(const float* __restrict__ in, __bf16* __restrict__ out, int n) {
  int i = blockIdx.x * 256 + threadIdx.x;
  if (i < n) out[i] = (__bf16)in[i];
}

// ---- GEMM: C[M,N] = A[M,K] * B[N,K]^T, bf16 WMMA, fp32 out -----------------
// block 256 thr (8 waves), tile 128M x 128N, wave = 64x32 (4x2 wmma tiles).
// Tiles streamed with GLOBAL_LOAD_ASYNC_TO_LDS_B128, double-buffered LDS.
__global__ void __launch_bounds__(256)
k_gemm_bf16_nt(const __bf16* __restrict__ A, const __bf16* __restrict__ B,
               float* __restrict__ C, int M, int N, int K) {
  __shared__ __bf16 sA[2][128 * 32];
  __shared__ __bf16 sB[2][128 * 32];
  const int tid  = threadIdx.x;
  const int lane = tid & 31;
  const int wave = tid >> 5;
  const int wm = wave >> 2;            // 0..1 -> 64-row strip
  const int wn = wave & 3;             // 0..3 -> 32-col strip
  const int m0 = blockIdx.y * 128;
  const int n0 = blockIdx.x * 128;

  // This thread's two 16B chunks of each 128x32 tile (512 chunks / 256 thr).
  // chunk e: row = e/4, col elements = (e&3)*8 (row stride = 32 elts = 64B).
  const int e0 = tid, e1 = tid + 256;
  int br0 = n0 + (e0 >> 2); if (br0 > N - 1) br0 = N - 1;  // clamp tail tile
  int br1 = n0 + (e1 >> 2); if (br1 > N - 1) br1 = N - 1;
  const __bf16* ga0 = A + (size_t)(m0 + (e0 >> 2)) * K + (e0 & 3) * 8;
  const __bf16* ga1 = A + (size_t)(m0 + (e1 >> 2)) * K + (e1 & 3) * 8;
  const __bf16* gb0 = B + (size_t)br0 * K + (e0 & 3) * 8;
  const __bf16* gb1 = B + (size_t)br1 * K + (e1 & 3) * 8;
  const unsigned lA = lds_off_of(&sA[0][0]);
  const unsigned lB = lds_off_of(&sB[0][0]);

  auto issue = [&](int buf, int k0) {
    async_ld_b128(lA + buf * 8192 + e0 * 16, ga0 + k0);
    async_ld_b128(lA + buf * 8192 + e1 * 16, ga1 + k0);
    async_ld_b128(lB + buf * 8192 + e0 * 16, gb0 + k0);
    async_ld_b128(lB + buf * 8192 + e1 * 16, gb1 + k0);
  };

  v8f acc[4][2] = {};
  issue(0, 0);
  wait_asynccnt0();
  __syncthreads();
  int buf = 0;
  for (int k0 = 0; k0 < K; k0 += 32) {
    if (k0 + 32 < K) issue(buf ^ 1, k0 + 32);   // prefetch next tile
    const __bf16* tA = sA[buf];
    const __bf16* tB = sB[buf];
    v16bf bfrag[2];
#pragma unroll
    for (int j = 0; j < 2; ++j)
      bfrag[j] = frag_ld(tB + (wn * 32 + j * 16) * 32, 32, lane);
#pragma unroll
    for (int i = 0; i < 4; ++i) {
      v16bf af = frag_ld(tA + (wm * 64 + i * 16) * 32, 32, lane);
      acc[i][0] = wmma_bf16(af, bfrag[0], acc[i][0]);
      acc[i][1] = wmma_bf16(af, bfrag[1], acc[i][1]);
    }
    wait_asynccnt0();
    __syncthreads();
    buf ^= 1;
  }
  const int l2 = lane >> 4, lc = lane & 15;
#pragma unroll
  for (int i = 0; i < 4; ++i)
#pragma unroll
    for (int j = 0; j < 2; ++j) {
      int col = n0 + wn * 32 + j * 16 + lc;
      if (col < N) {
#pragma unroll
        for (int rr = 0; rr < 8; ++rr) {
          int row = m0 + wm * 64 + i * 16 + rr + 8 * l2;
          C[(size_t)row * N + col] = acc[i][j][rr];
        }
      }
    }
}

// ---- dt = softplus(raw + bias), dA = -exp(A_log)*dt ------------------------
__global__ void __launch_bounds__(256)
k_dtda(const float* __restrict__ zx, const float* __restrict__ dt_bias,
       const float* __restrict__ A_log, float* __restrict__ dt,
       float* __restrict__ dA) {
  int i = blockIdx.x * 256 + threadIdx.x;
  if (i >= BL * NH) return;
  int h = i & (NH - 1);
  int row = i >> 5;
  float v  = zx[(size_t)row * DPROJ + (DPROJ - NH) + h] + dt_bias[h];
  float sp = v > 20.f ? v : log1pf(expf(v));
  dt[i] = sp;
  dA[i] = -expf(A_log[h]) * sp;
}

// ---- causal depthwise conv (K=4) + bias + SiLU on xBC slice ----------------
__global__ void __launch_bounds__(256)
k_conv(const float* __restrict__ zx, const float* __restrict__ w,
       const float* __restrict__ bias, float* __restrict__ out) {
  size_t i = (size_t)blockIdx.x * 256 + threadIdx.x;
  if (i >= (size_t)BL * CONVD) return;
  int    cch = (int)(i % CONVD);
  size_t row = i / CONVD;
  int    l   = (int)(row & (LD - 1));
  float acc = bias[cch];
#pragma unroll
  for (int k = 0; k < 4; ++k) {
    int ls = l - 3 + k;
    float xv = (ls >= 0) ? zx[(row - 3 + k) * DPROJ + DI + cch] : 0.f;
    acc += w[cch * 4 + k] * xv;
  }
  out[row * CONVD + cch] = acc / (1.f + expf(-acc));  // SiLU
}

// ---- SSD intra-chunk: G = C.B^T, Y_diag = (G*Lmat).xdt, states = B^T.xd2 ---
// block 128 thr = 4 waves = 4 heads; B/C/B^T tiles shared across heads.
__global__ void __launch_bounds__(128)
k_ssd_intra(const float* __restrict__ xbc, const float* __restrict__ dt,
            const float* __restrict__ dA, float* __restrict__ ybuf,
            float* __restrict__ states, float* __restrict__ acumg,
            float* __restrict__ csum) {
  __shared__ __bf16 sB [CK * DSZ];
  __shared__ __bf16 sC [CK * DSZ];
  __shared__ __bf16 sBT[DSZ * CK];
  __shared__ __bf16 sGL[4][16 * CK];
  __shared__ float  sAc[4][CK];
  __shared__ float  sDt[4][CK];
  __shared__ float  sDec[4][CK];

  const int lane = threadIdx.x & 31;
  const int wave = threadIdx.x >> 5;
  const int hg = blockIdx.x & 7;
  const int bc = blockIdx.x >> 3;       // b*NCH + c
  const int b  = bc >> 5;
  const int c  = bc & 31;
  const int h  = hg * 4 + wave;
  const int rowb = b * LD + c * CK;

  for (int idx = threadIdx.x; idx < CK * DSZ; idx += 128) {
    int s = idx >> 7, n = idx & (DSZ - 1);
    float bv = xbc[(size_t)(rowb + s) * CONVD + DI + n];
    float cv = xbc[(size_t)(rowb + s) * CONVD + DI + DSZ + n];
    sB[idx]          = (__bf16)bv;
    sBT[n * CK + s]  = (__bf16)bv;
    sC[idx]          = (__bf16)cv;
  }
  if (lane == 0) {                      // sequential cumsum (64 adds, cheap)
    float run = 0.f;
    for (int s = 0; s < CK; ++s) {
      run += dA[(size_t)(rowb + s) * NH + h];
      sAc[wave][s] = run;
      sDt[wave][s] = dt[(size_t)(rowb + s) * NH + h];
    }
  }
  __syncthreads();
  const float afin = sAc[wave][CK - 1];
  for (int s = lane; s < CK; s += 32)   // wave-local, no block barrier needed
    sDec[wave][s] = expf(afin - sAc[wave][s]) * sDt[wave][s];

  const float* xrow  = xbc + (size_t)rowb * CONVD + h * HD; // x[s][p]
  const float* sDtW  = sDt[wave];
  const float* sDecW = sDec[wave];
  const int l2 = lane >> 4, lc = lane & 15;

  // preload x-operand fragments (strided K, scaled): xdt (dt) & xd2 (dt*decay)
  v16bf xdf[2][4], xf[2][4];
#pragma unroll
  for (int ks = 0; ks < 2; ++ks)
#pragma unroll
    for (int pt = 0; pt < 4; ++pt) {
      xdf[ks][pt] = frag_ld_xk(xrow + ks * 32 * CONVD + pt * 16, CONVD,
                               sDtW + ks * 32, lane);
      xf [ks][pt] = frag_ld_xk(xrow + ks * 32 * CONVD + pt * 16, CONVD,
                               sDecW + ks * 32, lane);
    }

  // chunk states S[p][n] = sum_s x[s,p]*dt[s]*decay[s]*B[s,n]  (A = B^T)
  float* Sout = states + ((size_t)bc * NH + h) * (HD * DSZ);
#pragma unroll
  for (int nt = 0; nt < 8; ++nt) {
    v8f acc[4] = {};
#pragma unroll
    for (int ks = 0; ks < 2; ++ks) {
      v16bf aB = frag_ld(sBT + (nt * 16) * CK + ks * 32, CK, lane);
#pragma unroll
      for (int pt = 0; pt < 4; ++pt) acc[pt] = wmma_bf16(aB, xf[ks][pt], acc[pt]);
    }
#pragma unroll
    for (int pt = 0; pt < 4; ++pt)
#pragma unroll
      for (int rr = 0; rr < 8; ++rr) {
        int n = nt * 16 + rr + 8 * l2;
        int p = pt * 16 + lc;
        Sout[p * DSZ + n] = acc[pt][rr];
      }
  }

  // G strip -> mask/decay -> LDS relayout -> Y_diag strip
  for (int lt = 0; lt < 4; ++lt) {
    v8f g[4] = {};
#pragma unroll
    for (int kn = 0; kn < 4; ++kn) {
      v16bf aC = frag_ld(sC + (lt * 16) * DSZ + kn * 32, DSZ, lane);
#pragma unroll
      for (int st = 0; st < 4; ++st) {
        v16bf bB = frag_ld(sB + (st * 16) * DSZ + kn * 32, DSZ, lane);
        g[st] = wmma_bf16(aC, bB, g[st]);
      }
    }
#pragma unroll
    for (int st = 0; st < 4; ++st)
#pragma unroll
      for (int rr = 0; rr < 8; ++rr) {
        int l = lt * 16 + rr + 8 * l2;
        int s = st * 16 + lc;
        float v = (s <= l) ? g[st][rr] * expf(sAc[wave][l] - sAc[wave][s]) : 0.f;
        sGL[wave][(rr + 8 * l2) * CK + s] = (__bf16)v;
      }
    v8f y[4] = {};
#pragma unroll
    for (int ks = 0; ks < 2; ++ks) {
      v16bf aG = frag_ld(sGL[wave] + ks * 32, CK, lane);
#pragma unroll
      for (int pt = 0; pt < 4; ++pt) y[pt] = wmma_bf16(aG, xdf[ks][pt], y[pt]);
    }
#pragma unroll
    for (int pt = 0; pt < 4; ++pt)
#pragma unroll
      for (int rr = 0; rr < 8; ++rr) {
        int l = lt * 16 + rr + 8 * l2;
        int p = pt * 16 + lc;
        ybuf[(size_t)(rowb + l) * DI + h * HD + p] = y[pt][rr];
      }
  }

  for (int s = lane; s < CK; s += 32)
    acumg[((size_t)bc * NH + h) * CK + s] = sAc[wave][s];
  if (lane == 0) csum[(size_t)(b * NH + h) * NCH + c] = afin;
}

// ---- inter-chunk sequential scan over 32 chunks per (b,h) ------------------
__global__ void __launch_bounds__(256)
k_scan(float* __restrict__ states, const float* __restrict__ csum) {
  const int bh = blockIdx.x;            // b*NH + h
  const int b = bh >> 5, h = bh & 31;
  float run[32];
#pragma unroll
  for (int i = 0; i < 32; ++i) run[i] = 0.f;
  for (int c = 0; c < NCH; ++c) {
    const float dec = expf(csum[(size_t)bh * NCH + c]);
    float* S = states + ((size_t)(b * NCH + c) * NH + h) * (HD * DSZ);
#pragma unroll
    for (int i = 0; i < 32; ++i) {
      int e = i * 256 + threadIdx.x;    // coalesced
      float v = S[e];
      S[e] = run[i];                    // in-place: states -> states_in
      run[i] = run[i] * dec + v;
    }
  }
}

// ---- Y_off = exp(acum)*C.state_in^T, + Y_diag + x*D, gate by SiLU(z) -------
__global__ void __launch_bounds__(128)
k_ssd_inter(const float* __restrict__ xbc, const float* __restrict__ zx,
            const float* __restrict__ states, const float* __restrict__ acumg,
            const float* __restrict__ Dv, float* __restrict__ ybuf) {
  __shared__ __bf16 sC[CK * DSZ];
  __shared__ float  sAc[4][CK];
  const int lane = threadIdx.x & 31;
  const int wave = threadIdx.x >> 5;
  const int hg = blockIdx.x & 7;
  const int bc = blockIdx.x >> 3;
  const int b  = bc >> 5;
  const int c  = bc & 31;
  const int h  = hg * 4 + wave;
  const int rowb = b * LD + c * CK;

  for (int idx = threadIdx.x; idx < CK * DSZ; idx += 128) {
    int s = idx >> 7, n = idx & (DSZ - 1);
    sC[idx] = (__bf16)xbc[(size_t)(rowb + s) * CONVD + DI + DSZ + n];
  }
  for (int s = lane; s < CK; s += 32)
    sAc[wave][s] = acumg[((size_t)bc * NH + h) * CK + s];
  __syncthreads();

  const float* Sin = states + ((size_t)bc * NH + h) * (HD * DSZ); // [p][n]
  const float  Dh  = Dv[h];
  const int l2 = lane >> 4, lc = lane & 15;
  for (int lt = 0; lt < 4; ++lt) {
    v8f acc[4] = {};
#pragma unroll
    for (int kn = 0; kn < 4; ++kn) {
      v16bf aC = frag_ld(sC + (lt * 16) * DSZ + kn * 32, DSZ, lane);
#pragma unroll
      for (int pt = 0; pt < 4; ++pt) {
        v16bf bS = frag_ld_f32(Sin + (pt * 16) * DSZ + kn * 32, DSZ, lane);
        acc[pt] = wmma_bf16(aC, bS, acc[pt]);
      }
    }
#pragma unroll
    for (int pt = 0; pt < 4; ++pt)
#pragma unroll
      for (int rr = 0; rr < 8; ++rr) {
        int l = lt * 16 + rr + 8 * l2;
        int p = pt * 16 + lc;
        size_t t = (size_t)rowb + l;
        float yv = ybuf[t * DI + h * HD + p] + acc[pt][rr] * expf(sAc[wave][l]);
        yv += xbc[t * CONVD + h * HD + p] * Dh;
        float zv = zx[t * DPROJ + h * HD + p];
        yv *= zv / (1.f + expf(-zv));   // * SiLU(z)
        ybuf[t * DI + h * HD + p] = yv;
      }
  }
}

// ---- RMSNorm over DI, scale by norm_w, emit bf16 for out-proj GEMM ---------
__global__ void __launch_bounds__(256)
k_norm(const float* __restrict__ ybuf, const float* __restrict__ nw,
       __bf16* __restrict__ ybf) {
  __shared__ float sred[256];
  const int row = blockIdx.x;
  float ss = 0.f;
  for (int i = threadIdx.x; i < DI; i += 256) {
    float v = ybuf[(size_t)row * DI + i];
    ss += v * v;
  }
  sred[threadIdx.x] = ss;
  __syncthreads();
  for (int off = 128; off > 0; off >>= 1) {
    if (threadIdx.x < off) sred[threadIdx.x] += sred[threadIdx.x + off];
    __syncthreads();
  }
  const float scale = rsqrtf(sred[0] / (float)DI + 1e-5f);
  for (int i = threadIdx.x; i < DI; i += 256)
    ybf[(size_t)row * DI + i] = (__bf16)(ybuf[(size_t)row * DI + i] * scale * nw[i]);
}

// ---- launcher ---------------------------------------------------------------
extern "C" void kernel_launch(void* const* d_in, const int* in_sizes, int n_in,
                              void* d_out, int out_size, void* d_ws, size_t ws_size,
                              hipStream_t stream) {
  (void)in_sizes; (void)n_in; (void)out_size; (void)ws_size;
  const float* u      = (const float*)d_in[0];
  const float* W_in   = (const float*)d_in[1];
  const float* conv_w = (const float*)d_in[2];
  const float* conv_b = (const float*)d_in[3];
  const float* dt_b   = (const float*)d_in[4];
  const float* A_log  = (const float*)d_in[5];
  const float* Dv     = (const float*)d_in[6];
  const float* norm_w = (const float*)d_in[7];
  const float* W_out  = (const float*)d_in[8];
  float* out = (float*)d_out;

  char* ws = (char*)d_ws;
  size_t off = 0;
  auto alloc = [&](size_t bytes) -> void* {
    off = (off + 255) & ~(size_t)255;
    void* p = ws + off;
    off += bytes;
    return p;
  };
  float*  zx     = (float*) alloc((size_t)BL * DPROJ * 4);
  float*  xbca   = (float*) alloc((size_t)BL * CONVD * 4);
  float*  dt     = (float*) alloc((size_t)BL * NH * 4);
  float*  dA     = (float*) alloc((size_t)BL * NH * 4);
  float*  ybuf   = (float*) alloc((size_t)BL * DI * 4);
  float*  states = (float*) alloc((size_t)BD * NCH * NH * HD * DSZ * 4);
  float*  acum   = (float*) alloc((size_t)BD * NCH * NH * CK * 4);
  float*  csum   = (float*) alloc((size_t)BD * NH * NCH * 4);
  __bf16* ubf    = (__bf16*)alloc((size_t)BL * DM * 2);
  __bf16* wibf   = (__bf16*)alloc((size_t)DPROJ * DM * 2);
  __bf16* wobf   = (__bf16*)alloc((size_t)DM * DI * 2);
  __bf16* ybf    = (__bf16*)alloc((size_t)BL * DI * 2);

  k_f2bf<<<(BL * DM + 255) / 256, 256, 0, stream>>>(u, ubf, BL * DM);
  k_f2bf<<<(DPROJ * DM + 255) / 256, 256, 0, stream>>>(W_in, wibf, DPROJ * DM);
  k_f2bf<<<(DM * DI + 255) / 256, 256, 0, stream>>>(W_out, wobf, DM * DI);

  // in-proj: M=4096, N=4384 (35 col tiles of 128, tail guarded), K=1024
  k_gemm_bf16_nt<<<dim3((DPROJ + 127) / 128, BL / 128), 256, 0, stream>>>(
      ubf, wibf, zx, BL, DPROJ, DM);

  k_dtda<<<(BL * NH + 255) / 256, 256, 0, stream>>>(zx, dt_b, A_log, dt, dA);
  k_conv<<<(int)(((size_t)BL * CONVD + 255) / 256), 256, 0, stream>>>(
      zx, conv_w, conv_b, xbca);

  k_ssd_intra<<<BD * NCH * (NH / 4), 128, 0, stream>>>(xbca, dt, dA, ybuf,
                                                       states, acum, csum);
  k_scan<<<BD * NH, 256, 0, stream>>>(states, csum);
  k_ssd_inter<<<BD * NCH * (NH / 4), 128, 0, stream>>>(xbca, zx, states, acum,
                                                       Dv, ybuf);
  k_norm<<<BL, 256, 0, stream>>>(ybuf, norm_w, ybf);

  // out-proj: M=4096, N=1024, K=2048
  k_gemm_bf16_nt<<<dim3(DM / 128, BL / 128), 256, 0, stream>>>(
      ybf, wobf, out, BL, DM, DI);
}
